// Evoformer2bEncoder_2173253452658
// MI455X (gfx1250) — compile-verified, matrix operand order checked by reading, not run on hardware
//
#include <hip/hip_runtime.h>
#include <math.h>

// ---------------------------------------------------------------------------
// Evoformer2b encoder for MI455X (gfx1250, wave32, WMMA).
// Dims: NF=1, NLOC=512, NNEI=64, H=8, HEAD_DIM=128, FEAT=1024, FFN=2048, L=4.
// Bandwidth-bound (126MB fp32 weights @ 23.3TB/s ~ 5.4us floor); GEMM math in
// bf16 WMMA (f32 accum), gathered attention in VALU (negligible FLOPs).
// GEMM: 64x64 block tile, 8 waves (4Mx2N), 16x32 per wave (2 accumulators,
// shared A fragment), double-buffered LDS, one barrier per k-step.
// ---------------------------------------------------------------------------

typedef __bf16 bf16;
typedef __attribute__((ext_vector_type(16))) __bf16 v16bf;
typedef __attribute__((ext_vector_type(8)))  __bf16 bf16x8;
typedef __attribute__((ext_vector_type(8)))  float  v8f;

#define NLOC 512
#define NNEI 64
#define NHEAD 8
#define HDIM 128
#define FEAT 1024
#define FFN  2048
#define NLAYER 4

__device__ __forceinline__ v16bf frag_from_lds(const bf16* row, int k0) {
    bf16x8 lo = *(const bf16x8*)(row + k0);
    bf16x8 hi = *(const bf16x8*)(row + k0 + 16);
    return __builtin_shufflevector(lo, hi, 0,1,2,3,4,5,6,7,8,9,10,11,12,13,14,15);
}

// ---------------------------------------------------------------------------
// Tiled GEMM: C[M,N] = act(A[M,K] @ W[K,N] + bias[N] (+ res[M,N]))
// ACT: 0=none, 1=tanh, 2=exact gelu. Order: acc + bias (+res) then act.
// ---------------------------------------------------------------------------
template <int ACT, bool RES>
__global__ __launch_bounds__(256) void gemm_wmma_kernel(
    const float* __restrict__ A, const float* __restrict__ W,
    const float* __restrict__ bias, const float* __restrict__ res,
    float* __restrict__ C, int M, int N, int K)
{
    __shared__ bf16 As[2][64][40];   // [buf][m][k], row stride 80B (16B aligned)
    __shared__ bf16 Bs[2][64][40];   // [buf][n][k] (transposed from W)

    const int tid  = threadIdx.x;
    const int lane = tid & 31;
    const int wave = tid >> 5;
    const int subM = wave & 3;     // 0..3  -> rows subM*16..+16
    const int subN = wave >> 2;    // 0..1  -> cols subN*32..+32 (two 16-wide)
    const int tileN = blockIdx.x * 64;
    const int tileM = blockIdx.y * 64;

    const int l16 = lane & 15;
    const int hgr = lane >> 4;     // lane half-group: 0 or 1
    const int k0  = 8 * hgr;       // K pattern base per ISA 16-bit A/B layout

    // staging indices
    const int ar  = tid >> 2;          // A row 0..63
    const int ac0 = (tid & 3) * 8;     // A col base 0,8,16,24
    const int br  = tid >> 3;          // B k-row 0..31
    const int bc0 = (tid & 7) * 8;     // B n-col base 0..56

    v8f acc0 = {0.f,0.f,0.f,0.f,0.f,0.f,0.f,0.f};
    v8f acc1 = {0.f,0.f,0.f,0.f,0.f,0.f,0.f,0.f};

    // --- stage k-tile `kk` into buffer `b` ---
    auto stage = [&](int b, int kk) {
        {
            const float* src = A + (size_t)(tileM + ar) * K + kk + ac0;
            __builtin_prefetch(src + 64, 0, 1);   // -> global_prefetch_b8
#pragma unroll
            for (int e = 0; e < 8; ++e) As[b][ar][ac0 + e] = (bf16)src[e];
        }
        {
            const float* src = W + (size_t)(kk + br) * N + tileN + bc0;
            __builtin_prefetch(src + (size_t)32 * N, 0, 1);
#pragma unroll
            for (int e = 0; e < 8; ++e) Bs[b][bc0 + e][br] = (bf16)src[e];
        }
    };

    stage(0, 0);
    __syncthreads();

    int buf = 0;
    for (int kk = 0; kk < K; kk += 32) {
        if (kk + 32 < K) stage(buf ^ 1, kk + 32);   // overlap with WMMAs below

        v16bf afrag  = frag_from_lds(&As[buf][subM * 16 + l16][0], k0);
        v16bf bfrag0 = frag_from_lds(&Bs[buf][subN * 32 + l16][0], k0);
        v16bf bfrag1 = frag_from_lds(&Bs[buf][subN * 32 + 16 + l16][0], k0);

        acc0 = __builtin_amdgcn_wmma_f32_16x16x32_bf16(
            false, afrag, false, bfrag0, (short)0, acc0, false, false);
        acc1 = __builtin_amdgcn_wmma_f32_16x16x32_bf16(
            false, afrag, false, bfrag1, (short)0, acc1, false, false);

        __syncthreads();   // readers of buf done; writers of buf^1 done
        buf ^= 1;
    }

    // --- epilogue: C/D layout: VGPR r -> M = r + 8*hgr, N = l16 ---
    const int colg0 = tileN + subN * 32 + l16;
    const int colg1 = colg0 + 16;
    const float bv0 = bias[colg0];
    const float bv1 = bias[colg1];
#pragma unroll
    for (int r = 0; r < 8; ++r) {
        const int rowg = tileM + subM * 16 + r + 8 * hgr;
        float v0 = acc0[r] + bv0;
        float v1 = acc1[r] + bv1;
        if (RES) {
            v0 += res[(size_t)rowg * N + colg0];
            v1 += res[(size_t)rowg * N + colg1];
        }
        if (ACT == 1) { v0 = tanhf(v0); v1 = tanhf(v1); }
        else if (ACT == 2) {
            v0 = 0.5f * v0 * (1.0f + erff(v0 * 0.70710678118654752440f));
            v1 = 0.5f * v1 * (1.0f + erff(v1 * 0.70710678118654752440f));
        }
        C[(size_t)rowg * N + colg0] = v0;
        C[(size_t)rowg * N + colg1] = v1;
    }
}

// ---------------------------------------------------------------------------
// LayerNorm: y = (x - mean) * rsqrt(var + eps) * g + b, one row per block.
// ---------------------------------------------------------------------------
__global__ __launch_bounds__(256) void ln_kernel(
    const float* __restrict__ x, const float* __restrict__ g,
    const float* __restrict__ b, float* __restrict__ y)
{
    const int i = blockIdx.x, tid = threadIdx.x;
    __shared__ float red[32];
    float v[4], s = 0.f, s2 = 0.f;
#pragma unroll
    for (int e = 0; e < 4; ++e) {
        v[e] = x[(size_t)i * FEAT + tid + 256 * e];
        s += v[e]; s2 += v[e] * v[e];
    }
#pragma unroll
    for (int off = 16; off; off >>= 1) { s += __shfl_xor(s, off); s2 += __shfl_xor(s2, off); }
    const int w = tid >> 5, lane = tid & 31;
    if (lane == 0) { red[w] = s; red[8 + w] = s2; }
    __syncthreads();
    if (tid == 0) {
        float a = 0.f, a2 = 0.f;
        for (int k = 0; k < 8; ++k) { a += red[k]; a2 += red[8 + k]; }
        red[16] = a * (1.f / FEAT);
        red[17] = a2 * (1.f / FEAT);
    }
    __syncthreads();
    const float mean = red[16];
    const float var  = red[17] - mean * mean;
    const float rstd = rsqrtf(var + 1e-5f);
#pragma unroll
    for (int e = 0; e < 4; ++e) {
        const int c = tid + 256 * e;
        y[(size_t)i * FEAT + c] = (v[e] - mean) * rstd * g[c] + b[c];
    }
}

// ---------------------------------------------------------------------------
// Pair projection: pair[h,i,j] = pair_rep[i,j,:] @ Wp[:,h] + bp[h]
// input_pair stored raw; pair stored with -inf at masked positions.
// ---------------------------------------------------------------------------
__global__ __launch_bounds__(256) void pair_proj_kernel(
    const float* __restrict__ pair_rep, const float* __restrict__ wp,
    const float* __restrict__ bp, const unsigned char* __restrict__ mask,
    float* __restrict__ pair, float* __restrict__ input_pair)
{
    const int i = blockIdx.x, tid = threadIdx.x;
    __shared__ float wps[800];
    __shared__ float bps[8];
    for (int c = tid; c < 800; c += 256) wps[c] = wp[c];
    if (tid < 8) bps[tid] = bp[tid];
    __syncthreads();
    for (int p = tid; p < NNEI * NHEAD; p += 256) {
        const int j = p >> 3, h = p & 7;
        const float* src = pair_rep + ((size_t)i * NNEI + j) * 100;
        float a = bps[h];
        for (int c = 0; c < 100; ++c) a = fmaf(src[c], wps[c * 8 + h], a);
        const size_t idx = ((size_t)h * NLOC + i) * NNEI + j;
        input_pair[idx] = a;
        pair[idx] = mask[i * NNEI + j] ? a : -INFINITY;
    }
}

// ---------------------------------------------------------------------------
// Gathered attention, one query row per block (8 warps = 8 heads' softmax).
// scores = mask ? q.k : -inf ; scores += pair ; pair = scores (carry) ;
// o = softmax(scores) @ gathered V ; x += o (residual fused).
// ---------------------------------------------------------------------------
__global__ __launch_bounds__(256) void attn_kernel(
    const float* __restrict__ qkv, const int* __restrict__ nlist,
    const unsigned char* __restrict__ mask,
    float* __restrict__ pair, float* __restrict__ x)
{
    const int i = blockIdx.x, tid = threadIdx.x;
    __shared__ float qs[FEAT];
    __shared__ float attn_s[NHEAD][NNEI];
    __shared__ int   nb[NNEI];
    __shared__ unsigned char mk[NNEI];
    const float scaling = 0.08838834764831845f;  // 128^-0.5

    for (int c = tid; c < FEAT; c += 256) qs[c] = qkv[(size_t)i * 3 * FEAT + c] * scaling;
    if (tid < NNEI) { nb[tid] = nlist[i * NNEI + tid]; mk[tid] = mask[i * NNEI + tid]; }
    __syncthreads();

    for (int p = tid; p < NHEAD * NNEI; p += 256) {
        const int h = p >> 6, j = p & 63;
        const float* krow = qkv + (size_t)nb[j] * 3 * FEAT + FEAT + h * HDIM;
        const float* qh = qs + h * HDIM;
        float d = 0.f;
#pragma unroll 8
        for (int e = 0; e < HDIM; ++e) d = fmaf(qh[e], krow[e], d);
        float sc = mk[j] ? d : -INFINITY;
        const size_t pidx = ((size_t)h * NLOC + i) * NNEI + j;
        sc += pair[pidx];
        pair[pidx] = sc;
        attn_s[h][j] = sc;
    }
    __syncthreads();

    {   // softmax: warp w owns head w, 2 values per lane
        const int w = tid >> 5, lane = tid & 31;
        float v0 = attn_s[w][lane], v1 = attn_s[w][lane + 32];
        float m = fmaxf(v0, v1);
#pragma unroll
        for (int off = 16; off; off >>= 1) m = fmaxf(m, __shfl_xor(m, off));
        float e0 = __expf(v0 - m), e1 = __expf(v1 - m);
        float s = e0 + e1;
#pragma unroll
        for (int off = 16; off; off >>= 1) s += __shfl_xor(s, off);
        const float inv = 1.0f / s;
        attn_s[w][lane] = e0 * inv;
        attn_s[w][lane + 32] = e1 * inv;
    }
    __syncthreads();

    for (int c = tid; c < FEAT; c += 256) {
        const int h = c >> 7;
        float o = 0.f;
#pragma unroll 4
        for (int j = 0; j < NNEI; ++j)
            o = fmaf(attn_s[h][j], qkv[(size_t)nb[j] * 3 * FEAT + 2 * FEAT + c], o);
        x[(size_t)i * FEAT + c] += o;   // residual fused
    }
}

// ---------------------------------------------------------------------------
// Per-row norm_x loss: relu(|sqrt(sum x^2 + 1e-10) - 32| - 1) -> rowbuf[i]
// ---------------------------------------------------------------------------
__global__ __launch_bounds__(256) void norm_x_kernel(
    const float* __restrict__ x, float* __restrict__ rowbuf)
{
    const int i = blockIdx.x, tid = threadIdx.x;
    __shared__ float red[8];
    float s = 0.f;
#pragma unroll
    for (int e = 0; e < 4; ++e) {
        const float v = x[(size_t)i * FEAT + tid + 256 * e];
        s += v * v;
    }
#pragma unroll
    for (int off = 16; off; off >>= 1) s += __shfl_xor(s, off);
    if ((tid & 31) == 0) red[tid >> 5] = s;
    __syncthreads();
    if (tid == 0) {
        float a = 0.f;
        for (int k = 0; k < 8; ++k) a += red[k];
        const float norm = sqrtf(a + 1e-10f);
        rowbuf[i] = fmaxf(fabsf(norm - 32.0f) - 1.0f, 0.0f);
    }
}

// ---------------------------------------------------------------------------
// Pair / delta outputs + per-row masked-mean norm_delta.
// ---------------------------------------------------------------------------
__global__ __launch_bounds__(64) void pair_out_kernel(
    const float* __restrict__ pair, const float* __restrict__ input_pair,
    const unsigned char* __restrict__ mask,
    float* __restrict__ out_pair, float* __restrict__ out_delta,
    float* __restrict__ rowbuf)
{
    const int i = blockIdx.x, j = threadIdx.x;
    __shared__ float nd_s[NNEI];
    __shared__ float mk_s[NNEI];
    const int m = mask[i * NNEI + j] ? 1 : 0;
    float ss = 0.f;
#pragma unroll
    for (int h = 0; h < NHEAD; ++h) {
        const size_t idx = ((size_t)h * NLOC + i) * NNEI + j;
        const float pv = pair[idx];
        out_pair[idx] = pv;
        const float d = m ? (pv - input_pair[idx]) : 0.0f;
        out_delta[((size_t)i * NNEI + j) * NHEAD + h] = d;
        ss += d * d;
    }
    const float nd = fmaxf(fabsf(sqrtf(ss + 1e-10f) - 2.82842712474619009760f) - 1.0f, 0.0f);
    nd_s[j] = m ? nd : 0.0f;
    mk_s[j] = (float)m;
    __syncthreads();
    if (j == 0) {
        float sn = 0.f, sm = 0.f;
        for (int t = 0; t < NNEI; ++t) { sn += nd_s[t]; sm += mk_s[t]; }
        rowbuf[i] = sn / (1e-10f + sm);
    }
}

// ---------------------------------------------------------------------------
// Deterministic final reduce of the two 512-row loss buffers -> scalars.
// ---------------------------------------------------------------------------
__global__ __launch_bounds__(256) void final_reduce_kernel(
    const float* __restrict__ r0, const float* __restrict__ r1,
    float* __restrict__ o0, float* __restrict__ o1)
{
    __shared__ float s0[256], s1[256];
    const int tid = threadIdx.x;
    s0[tid] = r0[tid] + r0[tid + 256];
    s1[tid] = r1[tid] + r1[tid + 256];
    __syncthreads();
    for (int off = 128; off; off >>= 1) {
        if (tid < off) { s0[tid] += s0[tid + off]; s1[tid] += s1[tid + off]; }
        __syncthreads();
    }
    if (tid == 0) {
        *o0 = s0[0] * (1.0f / NLOC);
        *o1 = s1[0] * (1.0f / NLOC);
    }
}

// ---------------------------------------------------------------------------
// Host-side orchestration.
// ---------------------------------------------------------------------------
extern "C" void kernel_launch(void* const* d_in, const int* in_sizes, int n_in,
                              void* d_out, int out_size, void* d_ws, size_t ws_size,
                              hipStream_t stream)
{
    (void)in_sizes; (void)n_in; (void)out_size; (void)ws_size;

    // --- inputs (setup_inputs dict order, params flattened depth-first) ---
    const float* atomic_rep = (const float*)d_in[0];           // 512x1024
    const float* pair_rep   = (const float*)d_in[1];           // 512x64x100
    const int*   nlist      = (const int*)  d_in[2];           // 512x64
    /* nlist_type d_in[3] unused by the reference math */
    const unsigned char* nlist_mask = (const unsigned char*)d_in[4]; // 512x64 bool
    const float* w_in   = (const float*)d_in[5];
    const float* b_in   = (const float*)d_in[6];
    const float* w_out  = (const float*)d_in[7];
    const float* b_out  = (const float*)d_in[8];
    const float* w_pair = (const float*)d_in[9];
    const float* b_pair = (const float*)d_in[10];
    const float* fln_g  = (const float*)d_in[11];
    const float* fln_b  = (const float*)d_in[12];
    const float* ln1_g  = (const float*)d_in[13];
    const float* ln1_b  = (const float*)d_in[14];
    const float* attn_w = (const float*)d_in[15];
    const float* attn_b = (const float*)d_in[16];
    const float* ln2_g  = (const float*)d_in[17];
    const float* ln2_b  = (const float*)d_in[18];
    const float* fc1_w  = (const float*)d_in[19];
    const float* fc1_b  = (const float*)d_in[20];
    const float* fc2_w  = (const float*)d_in[21];
    const float* fc2_b  = (const float*)d_in[22];

    // --- workspace layout (floats) ---
    float* ws = (float*)d_ws;
    float* x      = ws;                             // 512*1024
    float* h      = x   + (size_t)NLOC * FEAT;      // 512*1024
    float* qkv    = h   + (size_t)NLOC * FEAT;      // 512*3072
    float* ffn    = qkv + (size_t)NLOC * 3 * FEAT;  // 512*2048
    float* pair   = ffn + (size_t)NLOC * FFN;       // 8*512*64
    float* ipair  = pair + (size_t)NHEAD * NLOC * NNEI;
    float* rbuf0  = ipair + (size_t)NHEAD * NLOC * NNEI; // 512
    float* rbuf1  = rbuf0 + NLOC;                        // 512

    // --- output layout (return-order concatenation) ---
    float* out_x     = (float*)d_out;                         // 512*1024 (final-LN x)
    float* out_trans = out_x     + (size_t)NLOC * FEAT;       // 512*1024
    float* out_pair  = out_trans + (size_t)NLOC * FEAT;       // 8*512*64
    float* out_delta = out_pair  + (size_t)NHEAD * NLOC * NNEI; // 512*64*8
    float* out_nx    = out_delta + (size_t)NLOC * NNEI * NHEAD; // scalar
    float* out_nd    = out_nx + 1;                              // scalar

    const dim3 blk(256);

    // x = tanh(atomic @ W_in + b_in)
    gemm_wmma_kernel<1, false><<<dim3(FEAT / 64, NLOC / 64), blk, 0, stream>>>(
        atomic_rep, w_in, b_in, nullptr, x, NLOC, FEAT, FEAT);

    // pair bias (+ input_pair copy, masked -inf)
    pair_proj_kernel<<<dim3(NLOC), blk, 0, stream>>>(
        pair_rep, w_pair, b_pair, nlist_mask, pair, ipair);

    for (int l = 0; l < NLAYER; ++l) {
        ln_kernel<<<dim3(NLOC), blk, 0, stream>>>(x, ln1_g + (size_t)l * FEAT,
                                                  ln1_b + (size_t)l * FEAT, h);
        gemm_wmma_kernel<0, false><<<dim3(3 * FEAT / 64, NLOC / 64), blk, 0, stream>>>(
            h, attn_w + (size_t)l * FEAT * 3 * FEAT, attn_b + (size_t)l * 3 * FEAT,
            nullptr, qkv, NLOC, 3 * FEAT, FEAT);
        attn_kernel<<<dim3(NLOC), blk, 0, stream>>>(qkv, nlist, nlist_mask, pair, x);
        ln_kernel<<<dim3(NLOC), blk, 0, stream>>>(x, ln2_g + (size_t)l * FEAT,
                                                  ln2_b + (size_t)l * FEAT, h);
        gemm_wmma_kernel<2, false><<<dim3(FFN / 64, NLOC / 64), blk, 0, stream>>>(
            h, fc1_w + (size_t)l * FEAT * FFN, fc1_b + (size_t)l * FFN,
            nullptr, ffn, NLOC, FFN, FEAT);
        gemm_wmma_kernel<0, true><<<dim3(FEAT / 64, NLOC / 64), blk, 0, stream>>>(
            ffn, fc2_w + (size_t)l * FFN * FEAT, fc2_b + (size_t)l * FEAT,
            x /*residual*/, x, NLOC, FEAT, FFN);
    }

    // norm_x from pre-final-LN x
    norm_x_kernel<<<dim3(NLOC), blk, 0, stream>>>(x, rbuf0);

    // final LN -> out_x
    ln_kernel<<<dim3(NLOC), blk, 0, stream>>>(x, fln_g, fln_b, out_x);

    // transformed = tanh(out_x @ W_out + b_out)
    gemm_wmma_kernel<1, false><<<dim3(FEAT / 64, NLOC / 64), blk, 0, stream>>>(
        out_x, w_out, b_out, nullptr, out_trans, NLOC, FEAT, FEAT);

    // pair / delta outputs + per-row norm_delta
    pair_out_kernel<<<dim3(NLOC), dim3(64), 0, stream>>>(
        pair, ipair, nlist_mask, out_pair, out_delta, rbuf1);

    // deterministic scalar reductions
    final_reduce_kernel<<<dim3(1), blk, 0, stream>>>(rbuf0, rbuf1, out_nx, out_nd);
}
